// MOELayer_7705171329366
// MI455X (gfx1250) — matrix operand order repaired and checked by reading, not compiled, not run on hardware
//
#include <hip/hip_runtime.h>
#include <cstdint>
#include <cstddef>

// Problem dims (fixed by the reference)
#define NTOK 8192
#define DDIM 2048
#define EEXP 16
#define HDIM 1024
#define FDIM 1024
#define TOPK 2

typedef __attribute__((ext_vector_type(16))) __bf16 v16bf;
typedef __attribute__((ext_vector_type(8)))  float  v8f;

union Frag { v16bf v; uint4 q[2]; };

#if __has_builtin(__builtin_amdgcn_sched_barrier)
#define SCHED_FENCE() __builtin_amdgcn_sched_barrier(0)
#else
#define SCHED_FENCE()
#endif

__device__ __forceinline__ unsigned short f2bf(float f) {
  unsigned int u = __builtin_bit_cast(unsigned int, f);
  unsigned int r = u + 0x7FFFu + ((u >> 16) & 1u);   // round-to-nearest-even
  return (unsigned short)(r >> 16);
}

__device__ __forceinline__ float gelu_exact(float x) {
  return 0.5f * x * (1.0f + erff(x * 0.70710678118654752440f));
}

// Async global -> LDS 16-byte copy, per lane. Tracked by ASYNCcnt.
__device__ __forceinline__ void async_cp16(unsigned lds_off, const void* gptr) {
  asm volatile("global_load_async_to_lds_b128 %0, %1, off"
               :: "v"(lds_off), "v"((unsigned long long)(size_t)gptr)
               : "memory");
}
__device__ __forceinline__ void wait_async_le3() {
  asm volatile("s_wait_asynccnt 0x3" ::: "memory");
}
__device__ __forceinline__ void wait_async_0() {
  asm volatile("s_wait_asynccnt 0x0" ::: "memory");
}

// ---------------------------------------------------------------------------
// init: zero expert counters and importance accumulator (in d_out tail)
// ---------------------------------------------------------------------------
__global__ void k_init(int* cnt, int* ecnt2, float* imp) {
  int t = threadIdx.x;
  if (t < EEXP) { cnt[t] = 0; ecnt2[t] = 0; imp[t] = 0.0f; }
}

// ---------------------------------------------------------------------------
// fp32 -> bf16 (flat, vectorized by 4)
// ---------------------------------------------------------------------------
__global__ void k_cvt4(const float* __restrict__ src,
                       unsigned short* __restrict__ dst, long long n4) {
  long long i = (long long)blockIdx.x * blockDim.x + threadIdx.x;
  if (i >= n4) return;
  float4 v = ((const float4*)src)[i];
  unsigned int lo = (unsigned int)f2bf(v.x) | ((unsigned int)f2bf(v.y) << 16);
  unsigned int hi = (unsigned int)f2bf(v.z) | ((unsigned int)f2bf(v.w) << 16);
  ((uint2*)dst)[i] = make_uint2(lo, hi);
}

// ---------------------------------------------------------------------------
// Transposing convert: src [R x C] fp32 (batched by blockIdx.z with stride R*C)
//                   -> dst [C x R] bf16. LDS-tiled 32x32.
// ---------------------------------------------------------------------------
__global__ void k_tcvt(const float* __restrict__ src,
                       unsigned short* __restrict__ dst, int R, int C) {
  __shared__ float tile[32][33];
  size_t eo = (size_t)blockIdx.z * (size_t)R * (size_t)C;
  int c0 = blockIdx.x * 32, r0 = blockIdx.y * 32;
#pragma unroll
  for (int j = 0; j < 32; j += 8)
    tile[threadIdx.y + j][threadIdx.x] =
        src[eo + (size_t)(r0 + threadIdx.y + j) * C + (c0 + threadIdx.x)];
  __syncthreads();
#pragma unroll
  for (int j = 0; j < 32; j += 8)
    dst[eo + (size_t)(c0 + threadIdx.y + j) * R + (r0 + threadIdx.x)] =
        f2bf(tile[threadIdx.x][threadIdx.y + j]);
}

// ---------------------------------------------------------------------------
// Router: one wave (32 lanes) per token. 16 expert dots + 1 gate dot over D.
// Butterfly reduce -> every lane holds all 17 sums. Softmax, top-2, sigmoid.
// ---------------------------------------------------------------------------
__global__ void k_router(const float* __restrict__ x, const float* __restrict__ Wr,
                         const float* __restrict__ Wg,
                         int* __restrict__ topki, float* __restrict__ topkv,
                         float* __restrict__ gvec,
                         int* __restrict__ cnt, float* __restrict__ imp) {
  const int wid = threadIdx.x >> 5, lane = threadIdx.x & 31;
  const int n = blockIdx.x * 8 + wid;
  const float* xr = x + (size_t)n * DDIM;

  float acc[EEXP + 1];
#pragma unroll
  for (int e = 0; e <= EEXP; e++) acc[e] = 0.0f;

  for (int d = lane; d < DDIM; d += 32) {
    float xv = xr[d];
    const float* wr = Wr + (size_t)d * EEXP;
#pragma unroll
    for (int e = 0; e < EEXP; e++) acc[e] += xv * wr[e];
    acc[EEXP] += xv * Wg[d];
  }
#pragma unroll
  for (int e = 0; e <= EEXP; e++) {
#pragma unroll
    for (int off = 16; off >= 1; off >>= 1)
      acc[e] += __shfl_xor(acc[e], off, 32);
  }

  // softmax over 16 (all lanes redundantly)
  float m = acc[0];
#pragma unroll
  for (int e = 1; e < EEXP; e++) m = fmaxf(m, acc[e]);
  float p[EEXP]; float s = 0.0f;
#pragma unroll
  for (int e = 0; e < EEXP; e++) { p[e] = expf(acc[e] - m); s += p[e]; }
  float inv = 1.0f / s;
#pragma unroll
  for (int e = 0; e < EEXP; e++) p[e] *= inv;

  if (lane < EEXP) atomicAdd(imp + lane, p[lane] * (1.0f / (float)NTOK));

  if (lane == 0) {
    int i0 = 0; float v0 = p[0];
#pragma unroll
    for (int e = 1; e < EEXP; e++) if (p[e] > v0) { v0 = p[e]; i0 = e; }
    int i1 = -1; float v1 = -1.0f;
#pragma unroll
    for (int e = 0; e < EEXP; e++)
      if (e != i0 && p[e] > v1) { v1 = p[e]; i1 = e; }
    topki[2 * n] = i0; topki[2 * n + 1] = i1;
    topkv[2 * n] = v0; topkv[2 * n + 1] = v1;
    gvec[n] = 1.0f / (1.0f + expf(-acc[EEXP]));
    atomicAdd(cnt + i0, 1);
    atomicAdd(cnt + i1, 1);
  }
}

// ---------------------------------------------------------------------------
// Exclusive scan of expert counts + "load" output
// ---------------------------------------------------------------------------
__global__ void k_scan(const int* __restrict__ cnt, int* __restrict__ ebase,
                       float* __restrict__ loadOut) {
  if (threadIdx.x == 0) {
    int run = 0;
    for (int i = 0; i < EEXP; i++) { ebase[i] = run; run += cnt[i]; }
    float inv = 1.0f / ((float)run + 1e-12f);
    for (int i = 0; i < EEXP; i++) loadOut[i] = (float)cnt[i] * inv;
  }
}

// ---------------------------------------------------------------------------
// Dispatch: build per-expert compacted token lists (perm) and gate values
// ---------------------------------------------------------------------------
__global__ void k_dispatch(const int* __restrict__ topki, const float* __restrict__ topkv,
                           const int* __restrict__ ebase, int* __restrict__ ecnt2,
                           int* __restrict__ perm, float* __restrict__ pgate) {
  int n = blockIdx.x * blockDim.x + threadIdx.x;
  if (n >= NTOK) return;
#pragma unroll
  for (int k = 0; k < TOPK; k++) {
    int e = topki[2 * n + k];
    int pos = atomicAdd(ecnt2 + e, 1);
    int slot = ebase[e] + pos;
    perm[slot] = n;
    pgate[slot] = topkv[2 * n + k];
  }
}

// ---------------------------------------------------------------------------
// Grouped/dense bf16 WMMA GEMM with double-buffered async LDS staging. MODE:
//  0: Hs = gelu(x @ W1s)                       (dense, store bf16)
//  1: y  = g .* (Hs @ W2s)                     (dense, store f32)
//  2: Hbuf = gate .* gelu(gather(x) @ W1[e])   (expert, store bf16)
//  3: y += scatter(Hbuf @ W2[e])               (expert, atomicAdd f32)
// Block: 256 thr = 8 waves; block tile 64x128, wave tile 16x64, K-step 32.
// A tile (64x32, row-major) and B tile (128 cols x 32 k, col-major) staged to
// LDS once per block via global_load_async_to_lds_b128, double buffered;
// fragments read from LDS as contiguous b128 pairs, grouped ahead of the
// WMMA block (sched_barrier) so s_wait_dscnt staggers instead of stalling.
// ---------------------------------------------------------------------------
template <int MODE>
__global__ __launch_bounds__(256, 2)
void k_gemm(const unsigned short* __restrict__ A,
            const unsigned short* __restrict__ Bw,
            const int* __restrict__ perm, const float* __restrict__ pgate,
            const float* __restrict__ gvec, const int* __restrict__ cnt,
            const int* __restrict__ ebase,
            unsigned short* __restrict__ outBf, float* __restrict__ outF,
            int Kdim, int Nd) {
  // [0,4K): A buf0   [4K,8K): A buf1   [8K,16K): B buf0   [16K,24K): B buf1
  __shared__ __align__(128) unsigned char smem[24576];

  const int tid  = threadIdx.x;
  const int lane = tid & 31, wid = tid >> 5;
  const int waveM = wid >> 1, waveN = wid & 1;
  const int h = lane >> 4, l15 = lane & 15;

  int e = 0, cntE = 0, base0 = 0;
  const int row0 = blockIdx.y * 64;
  if constexpr (MODE >= 2) {
    e = blockIdx.z;
    cntE = cnt[e];
    if (row0 >= cntE) return;       // uniform early exit (before any barrier)
    base0 = ebase[e];
  }
  const int ntile0 = blockIdx.x * 128;

  // ---- A staging source: thread -> row = tid>>2 (0..63), 16B chunk = tid&3
  const int arow = tid >> 2, ac = tid & 3;
  const unsigned short* aRowPtr;
  if constexpr (MODE == 2) {
    int slot = base0 + min(row0 + arow, cntE - 1);
    aRowPtr = A + (size_t)perm[slot] * (size_t)Kdim;
  } else if constexpr (MODE == 3) {
    int slot = base0 + min(row0 + arow, cntE - 1);
    aRowPtr = A + (size_t)slot * (size_t)Kdim;
  } else {
    aRowPtr = A + (size_t)(row0 + arow) * (size_t)Kdim;
  }
  const char* aSrc = (const char*)aRowPtr + ac * 16;
  const unsigned aDst = (unsigned)(arow * 64 + ac * 16);

  // ---- B staging source: tasks tid and tid+256 -> col = id>>2, chunk = id&3
  const char* bSrc[2]; unsigned bDst[2];
#pragma unroll
  for (int j = 0; j < 2; j++) {
    int id = tid + j * 256;
    int col = id >> 2, ch = id & 3;
    bSrc[j] = (const char*)(Bw + ((size_t)e * (size_t)Nd + (size_t)(ntile0 + col)) *
                                  (size_t)Kdim) + ch * 16;
    bDst[j] = (unsigned)(8192 + col * 64 + ch * 16);
  }

  const unsigned smemBase = (unsigned)(size_t)(&smem[0]);

  auto issue = [&](int k0, int sel) {
    async_cp16(smemBase + (unsigned)(sel * 4096) + aDst, aSrc + (size_t)k0 * 2);
#pragma unroll
    for (int j = 0; j < 2; j++)
      async_cp16(smemBase + (unsigned)(sel * 8192) + bDst[j], bSrc[j] + (size_t)k0 * 2);
  };

  v8f acc[4];
#pragma unroll
  for (int t = 0; t < 4; t++)
#pragma unroll
    for (int r = 0; r < 8; r++) acc[t][r] = 0.0f;

  const int nsteps = Kdim >> 5;
  issue(0, 0);

  int sel = 0;
  for (int s = 0; s < nsteps; s++, sel ^= 1) {
    if (s + 1 < nsteps) {
      issue((s + 1) << 5, sel ^ 1);   // prefetch next tile into other buffer
      wait_async_le3();               // in-order: retires current tile's 3 ops
    } else {
      wait_async_0();
    }
    __syncthreads();                  // all waves' current tile landed in LDS

    // A fragment: row-major tile; lane row = waveM*16 + l15,
    // K = h*16..h*16+15 -> 32B at h*32
    const unsigned char* pa = smem + sel * 4096 + (waveM * 16 + l15) * 64 + h * 32;
    Frag a;
    a.q[0] = *(const uint4*)(pa);
    a.q[1] = *(const uint4*)(pa + 16);

    // B fragments: col-major tile; lane col, K = h*16..h*16+15 -> 32B at h*32
    Frag b[4];
#pragma unroll
    for (int t = 0; t < 4; t++) {
      const unsigned char* pb = smem + 8192 + sel * 8192 +
                                (waveN * 64 + t * 16 + l15) * 64 + h * 32;
      b[t].q[0] = *(const uint4*)(pb);
      b[t].q[1] = *(const uint4*)(pb + 16);
    }

    SCHED_FENCE();  // keep all 10 ds_load_b128 ahead of the WMMA block

#pragma unroll
    for (int t = 0; t < 4; t++)
      acc[t] = __builtin_amdgcn_wmma_f32_16x16x32_bf16(
          false, a.v, false, b[t].v, (short)0, acc[t], false, false);

    SCHED_FENCE();

    __syncthreads();                  // consume-done before buffer is reused
  }

  // Epilogue. C layout: VGPR r -> M = r (lanes 0-15) / r+8 (lanes 16-31)
#pragma unroll
  for (int t = 0; t < 4; t++) {
    int col = ntile0 + waveN * 64 + t * 16 + l15;
#pragma unroll
    for (int r = 0; r < 8; r++) {
      int mrow = row0 + waveM * 16 + r + 8 * h;
      float v = acc[t][r];
      if constexpr (MODE == 0) {
        outBf[(size_t)mrow * Nd + col] = f2bf(gelu_exact(v));
      } else if constexpr (MODE == 1) {
        outF[(size_t)mrow * Nd + col] = gvec[mrow] * v;
      } else if constexpr (MODE == 2) {
        if (mrow < cntE) {
          int slot = base0 + mrow;
          outBf[(size_t)slot * Nd + col] = f2bf(gelu_exact(v) * pgate[slot]);
        }
      } else {
        if (mrow < cntE) {
          int tok = perm[base0 + mrow];
          atomicAdd(&outF[(size_t)tok * Nd + col], v);
        }
      }
    }
  }
}

// ---------------------------------------------------------------------------
extern "C" void kernel_launch(void* const* d_in, const int* in_sizes, int n_in,
                              void* d_out, int out_size, void* d_ws, size_t ws_size,
                              hipStream_t stream) {
  const float* x   = (const float*)d_in[0];
  const float* Wr  = (const float*)d_in[1];
  const float* Wg  = (const float*)d_in[2];
  const float* W1  = (const float*)d_in[3];
  const float* W2  = (const float*)d_in[4];
  const float* W1s = (const float*)d_in[5];
  const float* W2s = (const float*)d_in[6];

  float* y       = (float*)d_out;
  float* imp     = y + (size_t)NTOK * DDIM;
  float* loadOut = imp + EEXP;

  // Workspace carve-out
  char* ws = (char*)d_ws;
  size_t off = 0;
  auto alloc = [&](size_t bytes) -> void* {
    void* p = ws + off;
    off += (bytes + 255) & ~(size_t)255;
    return p;
  };
  unsigned short* xbf  = (unsigned short*)alloc((size_t)NTOK * DDIM * 2);
  unsigned short* w1e  = (unsigned short*)alloc((size_t)EEXP * DDIM * HDIM * 2);
  unsigned short* w2e  = (unsigned short*)alloc((size_t)EEXP * HDIM * DDIM * 2);
  unsigned short* w1s  = (unsigned short*)alloc((size_t)DDIM * FDIM * 2);
  unsigned short* w2s  = (unsigned short*)alloc((size_t)FDIM * DDIM * 2);
  unsigned short* hs   = (unsigned short*)alloc((size_t)NTOK * FDIM * 2);
  unsigned short* hbuf = (unsigned short*)alloc((size_t)NTOK * TOPK * HDIM * 2);
  int*   perm  = (int*)  alloc((size_t)NTOK * TOPK * 4);
  float* pg    = (float*)alloc((size_t)NTOK * TOPK * 4);
  float* gv    = (float*)alloc((size_t)NTOK * 4);
  int*   tki   = (int*)  alloc((size_t)NTOK * TOPK * 4);
  float* tkv   = (float*)alloc((size_t)NTOK * TOPK * 4);
  int*   cnt   = (int*)  alloc(64);
  int*   ebase = (int*)  alloc(64);
  int*   ecnt2 = (int*)  alloc(64);

  // 1) init counters + importance
  k_init<<<1, 64, 0, stream>>>(cnt, ecnt2, imp);

  // 2) x -> bf16
  {
    long long n4 = (long long)NTOK * DDIM / 4;
    k_cvt4<<<(unsigned)((n4 + 255) / 256), 256, 0, stream>>>(x, xbf, n4);
  }

  // 3-6) weights -> bf16, transposed to [Ncols x K]
  dim3 tb(32, 8);
  k_tcvt<<<dim3(HDIM / 32, DDIM / 32, EEXP), tb, 0, stream>>>(W1,  w1e, DDIM, HDIM);
  k_tcvt<<<dim3(DDIM / 32, HDIM / 32, EEXP), tb, 0, stream>>>(W2,  w2e, HDIM, DDIM);
  k_tcvt<<<dim3(FDIM / 32, DDIM / 32, 1),    tb, 0, stream>>>(W1s, w1s, DDIM, FDIM);
  k_tcvt<<<dim3(DDIM / 32, FDIM / 32, 1),    tb, 0, stream>>>(W2s, w2s, FDIM, DDIM);

  // 7) router
  k_router<<<NTOK / 8, 256, 0, stream>>>(x, Wr, Wg, tki, tkv, gv, cnt, imp);

  // 8) scan + load output
  k_scan<<<1, 32, 0, stream>>>(cnt, ebase, loadOut);

  // 9) dispatch
  k_dispatch<<<NTOK / 256, 256, 0, stream>>>(tki, tkv, ebase, ecnt2, perm, pg);

  // 10) shared up:  Hs = gelu(x @ W1s)              [K=D, N=F]
  k_gemm<0><<<dim3(FDIM / 128, NTOK / 64), 256, 0, stream>>>(
      xbf, w1s, nullptr, nullptr, nullptr, nullptr, nullptr, hs, nullptr, DDIM, FDIM);

  // 11) shared down: y = g .* (Hs @ W2s)            [K=F, N=D]
  k_gemm<1><<<dim3(DDIM / 128, NTOK / 64), 256, 0, stream>>>(
      hs, w2s, nullptr, nullptr, gv, nullptr, nullptr, nullptr, y, FDIM, DDIM);

  // 12) expert up:  Hbuf = gate .* gelu(gather(x) @ W1[e])   [K=D, N=H]
  k_gemm<2><<<dim3(HDIM / 128, NTOK / 64, EEXP), 256, 0, stream>>>(
      xbf, w1e, perm, pg, nullptr, cnt, ebase, hbuf, nullptr, DDIM, HDIM);

  // 13) expert down: y += scatter(Hbuf @ W2[e])     [K=H, N=D]
  k_gemm<3><<<dim3(DDIM / 128, NTOK / 64, EEXP), 256, 0, stream>>>(
      hbuf, w2e, perm, nullptr, nullptr, cnt, ebase, nullptr, y, HDIM, DDIM);

  (void)in_sizes; (void)n_in; (void)out_size; (void)ws_size;
}
